// GraphSage_85787676771075
// MI455X (gfx1250) — matrix operand order repaired and checked by reading, not compiled, not run on hardware
//
#include <hip/hip_runtime.h>
#include <hip/hip_bf16.h>

#define N_NODES 50000
#define N_EDGES 800000
#define N_GRAPHS 64
#define HSTRIDE 256      // row stride of activation buffer h (fp32)
#define MT_PER_BLK 5     // 16-row M-tiles per GEMM workgroup (80 rows)
#define SCAN_BLOCKS 196  // ceil(50000/256)

typedef __attribute__((ext_vector_type(16))) __bf16 bf16x16;
typedef __attribute__((ext_vector_type(8)))  __bf16 bf16x8;
typedef __attribute__((ext_vector_type(8)))  float  f32x8;

// ---- float -> bf16, round-to-nearest-even ----
__device__ inline __bf16 f2bf(float f) {
    unsigned u = __builtin_bit_cast(unsigned, f);
    unsigned r = (u + 0x7fffu + ((u >> 16) & 1u)) >> 16;
    unsigned short s = (unsigned short)r;
    return __builtin_bit_cast(__bf16, s);
}

// ================= input staging =================
// h[:, :128] = concat(x, pos)
__global__ void __launch_bounds__(256) k_concat(const float* __restrict__ x,
                                                const float* __restrict__ pos,
                                                float* __restrict__ h) {
    int idx = blockIdx.x * 256 + threadIdx.x;     // N*128 threads
    int i = idx >> 7, c = idx & 127;
    if (i >= N_NODES) return;
    float v = (c < 125) ? x[(size_t)i * 125 + c] : pos[(size_t)i * 3 + (c - 125)];
    h[(size_t)i * HSTRIDE + c] = v;
}

// ================= CSR build (edge list is layer-invariant: built once) ====
__global__ void __launch_bounds__(256) k_deg(const int* __restrict__ dst,
                                             int* __restrict__ deg) {
    int e = blockIdx.x * 256 + threadIdx.x;
    if (e < N_EDGES) atomicAdd(&deg[dst[e]], 1);
}

// block-local exclusive scan (256 elems) + block totals
__global__ void __launch_bounds__(256) k_scan1(const int* __restrict__ deg,
                                               int* __restrict__ excl,
                                               int* __restrict__ bsum) {
    __shared__ int sh[256];
    int i = blockIdx.x * 256 + threadIdx.x;
    int v = (i < N_NODES) ? deg[i] : 0;
    sh[threadIdx.x] = v;
    __syncthreads();
    for (int off = 1; off < 256; off <<= 1) {
        int t = (threadIdx.x >= (unsigned)off) ? sh[threadIdx.x - off] : 0;
        __syncthreads();
        sh[threadIdx.x] += t;
        __syncthreads();
    }
    if (i < N_NODES) excl[i] = sh[threadIdx.x] - v;
    if (threadIdx.x == 255) bsum[blockIdx.x] = sh[255];
}

// exclusive scan of the block totals (single block)
__global__ void __launch_bounds__(256) k_scan2(const int* __restrict__ bsum,
                                               int* __restrict__ boff) {
    __shared__ int sh[256];
    int t = threadIdx.x;
    int v = (t < SCAN_BLOCKS) ? bsum[t] : 0;
    sh[t] = v;
    __syncthreads();
    for (int off = 1; off < 256; off <<= 1) {
        int u = (t >= (unsigned)off) ? sh[t - off] : 0;
        __syncthreads();
        sh[t] += u;
        __syncthreads();
    }
    boff[t] = sh[t] - v;
}

__global__ void __launch_bounds__(256) k_scan3(const int* __restrict__ excl,
                                               const int* __restrict__ boff,
                                               int* __restrict__ rowptr) {
    int i = blockIdx.x * 256 + threadIdx.x;
    if (i < N_NODES) rowptr[i] = excl[i] + boff[i >> 8];
    if (i == N_NODES) rowptr[N_NODES] = N_EDGES;
}

__global__ void __launch_bounds__(256) k_fill(const int* __restrict__ src,
                                              const int* __restrict__ dst,
                                              const int* __restrict__ rowptr,
                                              int* __restrict__ cursor,
                                              int* __restrict__ csr_src) {
    int e = blockIdx.x * 256 + threadIdx.x;
    if (e >= N_EDGES) return;
    int d = dst[e];
    int p = atomicAdd(&cursor[d], 1);
    csr_src[rowptr[d] + p] = src[e];
}

// ================= fused mean-gather + bf16 pack ===========================
// One wave per node: register-accumulated neighbor sum (NO float atomics),
// then writes Acat[i] = [ bf16(mean) | bf16(h_i) ] ready for WMMA.
template <int F>
__global__ void __launch_bounds__(256) k_gather_acat(const float* __restrict__ h,
                                                     const int* __restrict__ rowptr,
                                                     const int* __restrict__ csr_src,
                                                     __bf16* __restrict__ acat) {
    constexpr int NF = F / 32;
    constexpr int K  = 2 * F;
    int wid  = (blockIdx.x * 256 + threadIdx.x) >> 5;
    int lane = threadIdx.x & 31;
    if (wid >= N_NODES) return;
    int beg = rowptr[wid], end = rowptr[wid + 1];

    float acc[NF];
#pragma unroll
    for (int t = 0; t < NF; ++t) acc[t] = 0.0f;

    for (int base = beg; base < end; base += 32) {
        int n = end - base;
        if (n > 32) n = 32;
        int s = (lane < n) ? csr_src[base + lane] : 0;
        for (int j = 0; j < n; ++j) {
            int sj = __shfl(s, j, 32);
            const float* hs = h + (size_t)sj * HSTRIDE;
#pragma unroll
            for (int t = 0; t < NF; ++t) acc[t] += hs[lane + 32 * t];
        }
    }

    float inv = 1.0f / fmaxf((float)(end - beg), 1.0f);
    __bf16* arow = acat + (size_t)wid * K;
    const float* hrow = h + (size_t)wid * HSTRIDE;
#pragma unroll
    for (int t = 0; t < NF; ++t) {
        arow[lane + 32 * t]     = f2bf(acc[t] * inv);
        arow[F + lane + 32 * t] = f2bf(hrow[lane + 32 * t]);
    }
}

// ================= weight pack: Wcat = [Wl | Wr] in bf16 ===================
__global__ void __launch_bounds__(256) k_build_wcat(const float* __restrict__ Wl,
                                                    const float* __restrict__ Wr,
                                                    __bf16* __restrict__ wcat,
                                                    int F) {
    int K = 2 * F;
    int idx = blockIdx.x * 256 + threadIdx.x;     // 256*K threads
    int o = idx / K, k = idx % K;
    if (o >= 256) return;
    float v = (k < F) ? Wl[(size_t)o * F + k] : Wr[(size_t)o * F + (k - F)];
    wcat[(size_t)o * K + k] = f2bf(v);
}

// ================= bf16 WMMA GEMM: out = relu(Acat @ Wcat^T + bias) ========
// 625 blocks x 8 waves; each wave: 5 M-tiles x 2 N-tiles (B reused 5x).
// A frag (16x32): lane m=l&15, g=l>>4; j<8 -> k=j+8g, j>=8 -> k=j+8+8g
// B frag (32x16): lane n=l&15, g=l>>4; j<8 -> k=j+8g, j>=8 -> k=j+8+8g
// C/D (16x16 f32): VGPR r -> M=r+8g, N=l&15
__global__ void __launch_bounds__(256) k_sage_gemm(const __bf16* __restrict__ A,
                                                   const __bf16* __restrict__ Wc,
                                                   const float* __restrict__ bias,
                                                   float* __restrict__ out,
                                                   int K) {
    int lane = threadIdx.x & 31;
    int wave = threadIdx.x >> 5;
    int m = lane & 15;
    int g = lane >> 4;
    int row0 = blockIdx.x * (16 * MT_PER_BLK);

    const __bf16* arow = A + (size_t)(row0 + m) * K;
    int col0 = wave * 32 + m;          // N index == lane&15 by layout
    int col1 = col0 + 16;
    const __bf16* w0 = Wc + (size_t)col0 * K;
    const __bf16* w1 = Wc + (size_t)col1 * K;

    f32x8 acc0[MT_PER_BLK] = {};
    f32x8 acc1[MT_PER_BLK] = {};

    for (int kk = 0; kk < K; kk += 32) {
        bf16x16 b0, b1;
        {
            bf16x8 p = *(const bf16x8*)(w0 + kk + 8 * g);
            bf16x8 q = *(const bf16x8*)(w0 + kk + 16 + 8 * g);
#pragma unroll
            for (int j = 0; j < 8; ++j) { b0[j] = p[j]; b0[j + 8] = q[j]; }
        }
        {
            bf16x8 p = *(const bf16x8*)(w1 + kk + 8 * g);
            bf16x8 q = *(const bf16x8*)(w1 + kk + 16 + 8 * g);
#pragma unroll
            for (int j = 0; j < 8; ++j) { b1[j] = p[j]; b1[j + 8] = q[j]; }
        }
#pragma unroll
        for (int t = 0; t < MT_PER_BLK; ++t) {
            bf16x16 a;
            const __bf16* ap = arow + (size_t)(t * 16) * K + kk;
            bf16x8 lo = *(const bf16x8*)(ap + 8 * g);
            bf16x8 hi = *(const bf16x8*)(ap + 16 + 8 * g);
#pragma unroll
            for (int j = 0; j < 8; ++j) { a[j] = lo[j]; a[j + 8] = hi[j]; }
            acc0[t] = __builtin_amdgcn_wmma_f32_16x16x32_bf16(false, a, false, b0,
                                                              (short)0, acc0[t], false, false);
            acc1[t] = __builtin_amdgcn_wmma_f32_16x16x32_bf16(false, a, false, b1,
                                                              (short)0, acc1[t], false, false);
        }
    }

    float bias0 = bias[col0];
    float bias1 = bias[col1];
#pragma unroll
    for (int t = 0; t < MT_PER_BLK; ++t) {
#pragma unroll
        for (int r = 0; r < 8; ++r) {
            int node = row0 + t * 16 + r + 8 * g;
            out[(size_t)node * HSTRIDE + col0] = fmaxf(acc0[t][r] + bias0, 0.0f);
            out[(size_t)node * HSTRIDE + col1] = fmaxf(acc1[t][r] + bias1, 0.0f);
        }
    }
}

// ================= global max pool (monotone-uint atomicMax) ===============
__global__ void __launch_bounds__(256) k_pool(const float* __restrict__ h,
                                              const int* __restrict__ batch,
                                              unsigned* __restrict__ genc) {
    int idx = blockIdx.x * 256 + threadIdx.x;     // N*256 threads
    int i = idx >> 8, f = idx & 255;
    if (i >= N_NODES) return;
    float v = h[(size_t)i * HSTRIDE + f];
    unsigned b = __builtin_bit_cast(unsigned, v);
    unsigned e = (b & 0x80000000u) ? ~b : (b | 0x80000000u);
    atomicMax(&genc[(size_t)batch[i] * 256 + f], e);
}

// ================= dense heads =============================================
__global__ void __launch_bounds__(256) k_lin(const unsigned* __restrict__ genc,
                                             const float* __restrict__ W,
                                             const float* __restrict__ b,
                                             float* __restrict__ out) {
    __shared__ float row[256];
    int gi = blockIdx.x, t = threadIdx.x;
    unsigned e = genc[(size_t)gi * 256 + t];
    unsigned bits = (e & 0x80000000u) ? (e & 0x7fffffffu) : ~e;
    row[t] = __builtin_bit_cast(float, bits);
    __syncthreads();
    const float* w = W + (size_t)t * 256;
    float acc = b[t];
    for (int k = 0; k < 256; ++k) acc += row[k] * w[k];
    out[(size_t)gi * 256 + t] = acc;
}

__global__ void __launch_bounds__(256) k_out(const float* __restrict__ gl,
                                             const float* __restrict__ W,
                                             const float* __restrict__ b,
                                             float* __restrict__ out) {
    __shared__ float row[256];
    int gi = blockIdx.x, t = threadIdx.x;
    row[t] = gl[(size_t)gi * 256 + t];
    __syncthreads();
    if (t < 10) {
        const float* w = W + (size_t)t * 256;
        float acc = b[t];
        for (int k = 0; k < 256; ++k) acc += row[k] * w[k];
        out[(size_t)gi * 10 + t] = acc;
    }
}

extern "C" void kernel_launch(void* const* d_in, const int* in_sizes, int n_in,
                              void* d_out, int out_size, void* d_ws, size_t ws_size,
                              hipStream_t stream) {
    (void)in_sizes; (void)n_in; (void)out_size; (void)ws_size;

    const float* x     = (const float*)d_in[0];
    const float* pos   = (const float*)d_in[1];
    const int*   ei    = (const int*)d_in[2];      // [2, E] flat
    const int*   batch = (const int*)d_in[3];
    const float* Wl[3] = {(const float*)d_in[4], (const float*)d_in[7], (const float*)d_in[10]};
    const float* bl[3] = {(const float*)d_in[5], (const float*)d_in[8], (const float*)d_in[11]};
    const float* Wr[3] = {(const float*)d_in[6], (const float*)d_in[9], (const float*)d_in[12]};
    const float* linW  = (const float*)d_in[13];
    const float* linb  = (const float*)d_in[14];
    const float* outW  = (const float*)d_in[15];
    const float* outb  = (const float*)d_in[16];

    // workspace carve-up (all chunk sizes are 16B multiples)
    char* ws = (char*)d_ws;
    float*    h       = (float*)ws;    ws += (size_t)N_NODES * HSTRIDE * sizeof(float);
    __bf16*   acat    = (__bf16*)ws;   ws += (size_t)N_NODES * 512 * sizeof(__bf16);
    __bf16*   wcat    = (__bf16*)ws;   ws += (size_t)256 * 512 * sizeof(__bf16);
    int*      deg     = (int*)ws;      ws += (size_t)N_NODES * sizeof(int);
    int*      excl    = (int*)ws;      ws += (size_t)N_NODES * sizeof(int);
    int*      bsum    = (int*)ws;      ws += 256 * sizeof(int);
    int*      boff    = (int*)ws;      ws += 256 * sizeof(int);
    int*      rowptr  = (int*)ws;      ws += (size_t)(N_NODES + 16) * sizeof(int);
    int*      cursor  = (int*)ws;      ws += (size_t)N_NODES * sizeof(int);
    int*      csr_src = (int*)ws;      ws += (size_t)N_EDGES * sizeof(int);
    unsigned* genc    = (unsigned*)ws; ws += (size_t)N_GRAPHS * 256 * sizeof(unsigned);
    float*    glin    = (float*)ws;    ws += (size_t)N_GRAPHS * 256 * sizeof(float);

    const int* src = ei;
    const int* dst = ei + N_EDGES;

    // stage inputs + build CSR once (edge structure is identical for all layers)
    k_concat<<<(N_NODES * 128) / 256, 256, 0, stream>>>(x, pos, h);
    hipMemsetAsync(deg, 0, (size_t)N_NODES * sizeof(int), stream);
    hipMemsetAsync(cursor, 0, (size_t)N_NODES * sizeof(int), stream);
    k_deg<<<(N_EDGES + 255) / 256, 256, 0, stream>>>(dst, deg);
    k_scan1<<<SCAN_BLOCKS, 256, 0, stream>>>(deg, excl, bsum);
    k_scan2<<<1, 256, 0, stream>>>(bsum, boff);
    k_scan3<<<SCAN_BLOCKS, 256, 0, stream>>>(excl, boff, rowptr);
    k_fill<<<(N_EDGES + 255) / 256, 256, 0, stream>>>(src, dst, rowptr, cursor, csr_src);

    for (int l = 0; l < 3; ++l) {
        int F = (l == 0) ? 128 : 256;
        int K = 2 * F;
        if (F == 128)
            k_gather_acat<128><<<(N_NODES * 32) / 256, 256, 0, stream>>>(h, rowptr, csr_src, acat);
        else
            k_gather_acat<256><<<(N_NODES * 32) / 256, 256, 0, stream>>>(h, rowptr, csr_src, acat);
        k_build_wcat<<<(256 * K) / 256, 256, 0, stream>>>(Wl[l], Wr[l], wcat, F);
        k_sage_gemm<<<N_NODES / (16 * MT_PER_BLK), 256, 0, stream>>>(acat, wcat, bl[l], h, K);
    }

    hipMemsetAsync(genc, 0, (size_t)N_GRAPHS * 256 * sizeof(unsigned), stream);
    k_pool<<<N_NODES, 256, 0, stream>>>(h, batch, genc);
    k_lin<<<N_GRAPHS, 256, 0, stream>>>(genc, linW, linb, glin);
    k_out<<<N_GRAPHS, 256, 0, stream>>>(glin, outW, outb, (float*)d_out);
}
